// SelfAtt_68736656605345
// MI455X (gfx1250) — matrix operand order repaired and checked by reading, not compiled, not run on hardware
//
#include <hip/hip_runtime.h>

typedef _Float16 half_t;
typedef __attribute__((ext_vector_type(16))) _Float16 v16h;
typedef __attribute__((ext_vector_type(8)))  _Float16 v8h;
typedef __attribute__((ext_vector_type(8)))  float    v8f;

static constexpr int BB = 64;
static constexpr int TT = 1024;
static constexpr int DD = 768;
static constexpr int AA = 128;

// ---------- WMMA helpers (layouts per CDNA5 ISA 7.12.2, wave32) ----------

__device__ __forceinline__ v16h cat8(v8h lo, v8h hi) {
  return __builtin_shufflevector(lo, hi, 0,1,2,3,4,5,6,7,8,9,10,11,12,13,14,15);
}

// A fragment, 16x32 f16, row-major source (ld in halfs), p = tile origin (m=0,k=0)
__device__ __forceinline__ v16h load_a_f16(const half_t* p, int ld, int lane) {
  int m = lane & 15, kb = (lane >> 4) * 8;
  const half_t* r = p + (size_t)m * ld + kb;
  v8h lo = *(const v8h*)(r);
  v8h hi = *(const v8h*)(r + 16);
  return cat8(lo, hi);
}

// A fragment with on-the-fly fp32 -> f16 conversion
__device__ __forceinline__ v16h load_a_f32(const float* p, int ld, int lane) {
  int m = lane & 15, kb = (lane >> 4) * 8;
  const float* r = p + (size_t)m * ld + kb;
  v8f lo = *(const v8f*)(r);
  v8f hi = *(const v8f*)(r + 16);
  return cat8(__builtin_convertvector(lo, v8h), __builtin_convertvector(hi, v8h));
}

// B fragment, 32x16 f16; source stored K-contiguous per column: element(k,n) at p + n*ld + k
__device__ __forceinline__ v16h load_b_f16(const half_t* p, int ld, int lane) {
  int n = lane & 15, kb = (lane >> 4) * 8;
  const half_t* r = p + (size_t)n * ld + kb;
  v8h lo = *(const v8h*)(r);
  v8h hi = *(const v8h*)(r + 16);
  return cat8(lo, hi);
}

__device__ __forceinline__ v8f wmma16(v16h a, v16h b, v8f c) {
  return __builtin_amdgcn_wmma_f32_16x16x32_f16(false, a, false, b, (short)0, c, false, false);
}

// ---------- K0a: x [B,T,D] fp32 -> xT [B,D,T] f16 (LDS-tiled transpose) ----------

__global__ void __launch_bounds__(256)
sa_xpose_kernel(const float* __restrict__ x, half_t* __restrict__ xT) {
  __shared__ half_t tile[32][33];
  int b  = blockIdx.z;
  int t0 = blockIdx.x * 32;          // T/32 = 32
  int d0 = blockIdx.y * 32;          // D/32 = 24
  int tx = threadIdx.x;              // 0..31
  int ty = threadIdx.y;              // 0..7
  const float* xb = x + (size_t)b * TT * DD;
  #pragma unroll
  for (int i = 0; i < 32; i += 8)
    tile[ty + i][tx] = (half_t)xb[(size_t)(t0 + ty + i) * DD + d0 + tx];
  __syncthreads();
  half_t* xTb = xT + (size_t)b * DD * TT;
  #pragma unroll
  for (int i = 0; i < 32; i += 8)
    xTb[(size_t)(d0 + ty + i) * TT + t0 + tx] = tile[tx][ty + i];
}

// ---------- K0b: W1 [D,A] -> W1t [A,D] f16 ; W2 [A,T] -> W2t [T,A] f16 ----------

__global__ void __launch_bounds__(256)
sa_wconv_kernel(const float* __restrict__ W1, const float* __restrict__ W2,
                half_t* __restrict__ W1t, half_t* __restrict__ W2t) {
  int i = blockIdx.x * 256 + threadIdx.x;
  if (i < AA * DD) { int a = i / DD, d = i - a * DD; W1t[i] = (half_t)W1[(size_t)d * AA + a]; }
  if (i < TT * AA) { int s = i / AA, a = i - s * AA; W2t[i] = (half_t)W2[(size_t)a * TT + s]; }
}

// ---------- K1: OA[r,a] = tanh(x[r,:] @ W1[:,a]) ; wave = 16 rows x 128 cols ----------
// Ping-pong double buffered: loads for stage i+1 are live across the WMMA burst of
// stage i, forcing distinct registers -> staggered loadcnt waits instead of 0x0.

__global__ void __launch_bounds__(256)
sa_gemm1_kernel(const float* __restrict__ x, const half_t* __restrict__ W1t,
                half_t* __restrict__ OA) {
  int wave = blockIdx.x * 8 + (threadIdx.x >> 5);   // 4096 waves, 16 rows each
  int lane = threadIdx.x & 31;
  int m0 = wave * 16;
  const float* xr = x + (size_t)m0 * DD;
  v8f c[8] = {};
  v16h a0 = load_a_f32(xr, DD, lane);
  v16h b0[8];
  #pragma unroll
  for (int j = 0; j < 8; ++j) b0[j] = load_b_f16(W1t + (size_t)(j * 16) * DD, DD, lane);
  for (int k0 = 0; k0 < DD; k0 += 64) {          // 12 iterations, 2 k-steps each
    int k1 = k0 + 32;                            // always < DD
    v16h a1 = load_a_f32(xr + k1, DD, lane);
    v16h b1[8];
    #pragma unroll
    for (int j = 0; j < 8; ++j) b1[j] = load_b_f16(W1t + (size_t)(j * 16) * DD + k1, DD, lane);
    #pragma unroll
    for (int j = 0; j < 8; ++j) c[j] = wmma16(a0, b0[j], c[j]);
    int k2 = (k0 + 64 < DD) ? k0 + 64 : 0;       // branch-free wrap (harmless reload)
    a0 = load_a_f32(xr + k2, DD, lane);
    #pragma unroll
    for (int j = 0; j < 8; ++j) b0[j] = load_b_f16(W1t + (size_t)(j * 16) * DD + k2, DD, lane);
    #pragma unroll
    for (int j = 0; j < 8; ++j) c[j] = wmma16(a1, b1[j], c[j]);
  }
  int mr = (lane >> 4) * 8, nc = lane & 15;
  #pragma unroll
  for (int j = 0; j < 8; ++j) {
    #pragma unroll
    for (int v = 0; v < 8; ++v) {
      float t = fminf(fmaxf(c[j][v], -15.f), 15.f);   // tanh via exp, clamped (saturated anyway)
      float e = __expf(2.f * t);
      OA[(size_t)(m0 + mr + v) * AA + j * 16 + nc] = (half_t)((e - 1.f) / (e + 1.f));
    }
  }
}

// ---------- K2: St[b][s][t] = sum_a W2t[s,a] * OA[b][t,a] ; wave = 32x32 tile ----------

__global__ void __launch_bounds__(256)
sa_gemm2_kernel(const half_t* __restrict__ W2t, const half_t* __restrict__ OA,
                half_t* __restrict__ S) {
  int wave = blockIdx.x * 8 + (threadIdx.x >> 5);   // 64 * 1024 waves
  int lane = threadIdx.x & 31;
  int b    = wave >> 10;
  int w    = wave & 1023;
  int s0 = (w >> 5) * 32;
  int t0 = (w & 31) * 32;
  const half_t* OAb = OA + (size_t)b * TT * AA;
  v8f c[2][2] = {};
  #pragma unroll
  for (int k0 = 0; k0 < AA; k0 += 32) {
    v16h a0 = load_a_f16(W2t + (size_t)s0 * AA + k0, AA, lane);
    v16h a1 = load_a_f16(W2t + (size_t)(s0 + 16) * AA + k0, AA, lane);
    v16h b0 = load_b_f16(OAb + (size_t)t0 * AA + k0, AA, lane);
    v16h b1 = load_b_f16(OAb + (size_t)(t0 + 16) * AA + k0, AA, lane);
    c[0][0] = wmma16(a0, b0, c[0][0]);
    c[0][1] = wmma16(a0, b1, c[0][1]);
    c[1][0] = wmma16(a1, b0, c[1][0]);
    c[1][1] = wmma16(a1, b1, c[1][1]);
  }
  half_t* Sb = S + (size_t)b * TT * TT;
  int mr = (lane >> 4) * 8, nc = lane & 15;
  #pragma unroll
  for (int mi = 0; mi < 2; ++mi)
    #pragma unroll
    for (int ni = 0; ni < 2; ++ni)
      #pragma unroll
      for (int v = 0; v < 8; ++v)
        Sb[(size_t)(s0 + mi * 16 + mr + v) * TT + t0 + ni * 16 + nc] = (half_t)c[mi][ni][v];
}

// ---------- K3: in-place row softmax of St (row = fixed (b,s), over t) ----------

__global__ void __launch_bounds__(256)
sa_softmax_kernel(half_t* __restrict__ S) {
  int row  = blockIdx.x * 8 + (threadIdx.x >> 5);   // B*T rows, one wave each
  int lane = threadIdx.x & 31;
  half_t* r = S + (size_t)row * TT + lane * 32;     // 32 halfs per lane, contiguous per wave
  v8f f[4];
  #pragma unroll
  for (int j = 0; j < 4; ++j)
    f[j] = __builtin_convertvector(*(const v8h*)(r + j * 8), v8f);
  float m = -3.0e38f;
  #pragma unroll
  for (int j = 0; j < 4; ++j)
    #pragma unroll
    for (int i = 0; i < 8; ++i) m = fmaxf(m, f[j][i]);
  #pragma unroll
  for (int off = 16; off > 0; off >>= 1) m = fmaxf(m, __shfl_xor(m, off, 32));
  float sum = 0.f;
  #pragma unroll
  for (int j = 0; j < 4; ++j)
    #pragma unroll
    for (int i = 0; i < 8; ++i) { float e = __expf(f[j][i] - m); f[j][i] = e; sum += e; }
  #pragma unroll
  for (int off = 16; off > 0; off >>= 1) sum += __shfl_xor(sum, off, 32);
  float inv = 1.f / sum;
  #pragma unroll
  for (int j = 0; j < 4; ++j) {
    #pragma unroll
    for (int i = 0; i < 8; ++i) f[j][i] *= inv;
    *(v8h*)(r + j * 8) = __builtin_convertvector(f[j], v8h);
  }
}

// ---------- K4: ot[b][s][d] = sum_t P[b][s][t] * xT[b][d][t] ; wave = 32x64 tile ----------
// Same ping-pong double buffering as K1.

__global__ void __launch_bounds__(256)
sa_gemm3_kernel(const half_t* __restrict__ P, const half_t* __restrict__ xT,
                float* __restrict__ out) {
  int wave = blockIdx.x * 8 + (threadIdx.x >> 5);   // 64 * 384 waves
  int lane = threadIdx.x & 31;
  int b = wave / 384;
  int w = wave - b * 384;
  int s0 = (w / 12) * 32;
  int d0 = (w - (w / 12) * 12) * 64;
  const half_t* Pb  = P  + (size_t)b * TT * TT;
  const half_t* xTb = xT + (size_t)b * DD * TT;
  v8f c[2][4] = {};
  v16h pa0 = load_a_f16(Pb + (size_t)s0 * TT, TT, lane);
  v16h pa1 = load_a_f16(Pb + (size_t)(s0 + 16) * TT, TT, lane);
  v16h pb[4];
  #pragma unroll
  for (int ni = 0; ni < 4; ++ni) pb[ni] = load_b_f16(xTb + (size_t)(d0 + ni * 16) * TT, TT, lane);
  for (int k0 = 0; k0 < TT; k0 += 64) {          // 16 iterations, 2 k-steps each
    int k1 = k0 + 32;                            // always < TT
    v16h qa0 = load_a_f16(Pb + (size_t)s0 * TT + k1, TT, lane);
    v16h qa1 = load_a_f16(Pb + (size_t)(s0 + 16) * TT + k1, TT, lane);
    v16h qb[4];
    #pragma unroll
    for (int ni = 0; ni < 4; ++ni)
      qb[ni] = load_b_f16(xTb + (size_t)(d0 + ni * 16) * TT + k1, TT, lane);
    #pragma unroll
    for (int ni = 0; ni < 4; ++ni) {
      c[0][ni] = wmma16(pa0, pb[ni], c[0][ni]);
      c[1][ni] = wmma16(pa1, pb[ni], c[1][ni]);
    }
    int k2 = (k0 + 64 < TT) ? k0 + 64 : 0;       // branch-free wrap (harmless reload)
    pa0 = load_a_f16(Pb + (size_t)s0 * TT + k2, TT, lane);
    pa1 = load_a_f16(Pb + (size_t)(s0 + 16) * TT + k2, TT, lane);
    #pragma unroll
    for (int ni = 0; ni < 4; ++ni)
      pb[ni] = load_b_f16(xTb + (size_t)(d0 + ni * 16) * TT + k2, TT, lane);
    #pragma unroll
    for (int ni = 0; ni < 4; ++ni) {
      c[0][ni] = wmma16(qa0, qb[ni], c[0][ni]);
      c[1][ni] = wmma16(qa1, qb[ni], c[1][ni]);
    }
  }
  float* outb = out + (size_t)b * TT * DD;
  int mr = (lane >> 4) * 8, nc = lane & 15;
  #pragma unroll
  for (int mi = 0; mi < 2; ++mi)
    #pragma unroll
    for (int ni = 0; ni < 4; ++ni)
      #pragma unroll
      for (int v = 0; v < 8; ++v)
        __builtin_nontemporal_store(c[mi][ni][v],
            outb + (size_t)(s0 + mi * 16 + mr + v) * DD + d0 + ni * 16 + nc);
}

// ---------- launch ----------

extern "C" void kernel_launch(void* const* d_in, const int* in_sizes, int n_in,
                              void* d_out, int out_size, void* d_ws, size_t ws_size,
                              hipStream_t stream) {
  const float* x  = (const float*)d_in[0];   // [B,T,D]
  const float* W1 = (const float*)d_in[1];   // [D,A]
  const float* W2 = (const float*)d_in[2];   // [A,T]
  float* out = (float*)d_out;                // [B,T,D] fp32

  half_t* ws  = (half_t*)d_ws;
  size_t  o   = 0;
  half_t* xT  = ws + o; o += (size_t)BB * DD * TT;   // 100.7 MB
  half_t* W1t = ws + o; o += (size_t)AA * DD;        // 0.2 MB
  half_t* W2t = ws + o; o += (size_t)TT * AA;        // 0.25 MB
  half_t* OA  = ws + o; o += (size_t)BB * TT * AA;   // 16.8 MB
  half_t* S   = ws + o;                              // 134 MB  (also holds P in-place)

  sa_xpose_kernel  <<<dim3(TT / 32, DD / 32, BB), dim3(32, 8), 0, stream>>>(x, xT);
  sa_wconv_kernel  <<<(TT * AA) / 256, 256, 0, stream>>>(W1, W2, W1t, W2t);
  sa_gemm1_kernel  <<<(BB * TT / 16) / 8, 256, 0, stream>>>(x, W1t, OA);
  sa_gemm2_kernel  <<<(BB * (TT / 32) * (TT / 32)) / 8, 256, 0, stream>>>(W2t, OA, S);
  sa_softmax_kernel<<<(BB * TT) / 8, 256, 0, stream>>>(S);
  sa_gemm3_kernel  <<<(BB * (TT / 32) * (DD / 64)) / 8, 256, 0, stream>>>(S, xT, out);
}
